// Qwen3_5FullAttention_11802570129919
// MI455X (gfx1250) — matrix operand order repaired and checked by reading, not compiled, not run on hardware
//
#include <hip/hip_runtime.h>
#include <math.h>

// ---------------------------------------------------------------------------
// Qwen3.5 full attention layer for gfx1250 (MI455X), bf16 WMMA pipeline.
// T=2048, HID=2048, H=16, KV=2, D=256, ROT=64, THETA=1e7, EPS=1e-6, SCALE=1/16
// ---------------------------------------------------------------------------

typedef __bf16 v16bf __attribute__((ext_vector_type(16)));
typedef __bf16 v2bf  __attribute__((ext_vector_type(2)));
typedef float  v8f   __attribute__((ext_vector_type(8)));
typedef float  v2f   __attribute__((ext_vector_type(2)));

union Frag {
    unsigned int u[8];
    v16bf        v;
};

__device__ __forceinline__ unsigned short f2bf(float f) {
    __bf16 h = (__bf16)f;                     // hardware cvt, round-to-nearest-even
    return __builtin_bit_cast(unsigned short, h);
}

// Packed f32x2 -> bf16x2 (lowers to v_cvt_pk_bf16_f32).
__device__ __forceinline__ unsigned int pk2bf(float a, float b) {
    v2f x = {a, b};
    v2bf h = __builtin_convertvector(x, v2bf);
    return __builtin_bit_cast(unsigned int, h);
}

// Truncate a generic pointer to a __shared__ object down to its LDS byte offset
// (flat LDS aperture keeps the LDS address in addr[31:0]).
__device__ __forceinline__ unsigned lds_off(const void* p) {
    return (unsigned)(uintptr_t)p;
}

// CDNA5 async global->LDS copy (16 bytes per lane), tracked on ASYNCcnt.
__device__ __forceinline__ void async_copy16(unsigned lds_byte_addr, const void* gaddr) {
    asm volatile("global_load_async_to_lds_b128 %0, %1, off"
                 :: "v"(lds_byte_addr), "v"(gaddr) : "memory");
}
__device__ __forceinline__ void async_wait0() {
    asm volatile("s_wait_asynccnt 0x0" ::: "memory");
}

// ---------------------------------------------------------------------------
// Generic GEMM: C[M x N] = X[M x K] * W[N x K]^T  (f32 in, f32 out)
// Tiles staged into LDS as bf16; compute via v_wmma_f32_16x16x32_bf16.
// Block tile: BM=128 x BN=64, BK=64. 256 threads = 8 waves (4 along M, 2 along N),
// each wave computes a 32x32 tile = 2x2 WMMA tiles.
// ---------------------------------------------------------------------------
#define BM 128
#define BN 64
#define BK 64

__global__ __launch_bounds__(256) void gemm_xwT(
    const float* __restrict__ X, const float* __restrict__ W,
    float* __restrict__ C, int Kdim, int N)
{
    __shared__ alignas(16) unsigned short lA[BM * BK];   // [m][k] row-major
    __shared__ alignas(16) unsigned short lB[BN * BK];   // [n][k] row-major

    const int tid   = threadIdx.x;
    const int lane  = tid & 31;
    const int wave  = tid >> 5;
    const int waveM = wave & 3;              // 0..3
    const int waveN = wave >> 2;             // 0..1
    const int m0    = blockIdx.x * BM;
    const int n0    = blockIdx.y * BN;
    const int g     = lane >> 4;             // lane half
    const int mr    = lane & 15;
    const int nc    = lane & 15;

    v8f acc[2][2] = {};

    for (int k0 = 0; k0 < Kdim; k0 += BK) {
        // Stage A tile: 2048 float4 loads, 8 per thread; packed bf16 stores
        for (int i = 0; i < 8; ++i) {
            int idx = tid + i * 256;
            int r = idx >> 4, c4 = idx & 15;
            const float4 xv = *(const float4*)(X + (size_t)(m0 + r) * Kdim + k0 + c4 * 4);
            *(uint2*)&lA[r * BK + c4 * 4] =
                make_uint2(pk2bf(xv.x, xv.y), pk2bf(xv.z, xv.w));
        }
        // Stage B tile: 1024 float4 loads, 4 per thread
        for (int i = 0; i < 4; ++i) {
            int idx = tid + i * 256;
            int r = idx >> 4, c4 = idx & 15;
            const float4 wv = *(const float4*)(W + (size_t)(n0 + r) * Kdim + k0 + c4 * 4);
            *(uint2*)&lB[r * BK + c4 * 4] =
                make_uint2(pk2bf(wv.x, wv.y), pk2bf(wv.z, wv.w));
        }
        __syncthreads();

        for (int s = 0; s < 2; ++s) {        // two K=32 steps per BK=64 tile
            Frag a[2], b[2];
            for (int v = 0; v < 8; ++v) {
                int kbA = s * 32 + ((v & 3) * 2) + ((v >> 2) * 16) + g * 8;
                int kbB = s * 32 + ((v & 3) * 2) + ((v >> 2) * 16) + g * 8;
                for (int i = 0; i < 2; ++i)
                    a[i].u[v] = *(const unsigned int*)&lA[(waveM * 32 + i * 16 + mr) * BK + kbA];
                for (int j = 0; j < 2; ++j)
                    b[j].u[v] = *(const unsigned int*)&lB[(waveN * 32 + j * 16 + nc) * BK + kbB];
            }
            for (int i = 0; i < 2; ++i)
                for (int j = 0; j < 2; ++j)
                    acc[i][j] = __builtin_amdgcn_wmma_f32_16x16x32_bf16(
                        false, a[i].v, false, b[j].v, (short)0, acc[i][j], false, false);
        }
        __syncthreads();
    }

    for (int i = 0; i < 2; ++i)
        for (int j = 0; j < 2; ++j)
            for (int r = 0; r < 8; ++r) {
                int m = m0 + waveM * 32 + i * 16 + r + 8 * g;
                int n = n0 + waveN * 32 + j * 16 + nc;
                C[(size_t)m * N + n] = acc[i][j][r];
            }
}

// ---------------------------------------------------------------------------
// RMSNorm + partial RoPE over D=256 (ROT=64). One block per (t, head) row.
// src f32 -> dst bf16 (ushort).
// ---------------------------------------------------------------------------
__global__ __launch_bounds__(256) void norm_rope(
    const int* __restrict__ pos, const float* __restrict__ src,
    const float* __restrict__ w, unsigned short* __restrict__ dst,
    int sStride, int sHead, int dStride, int dHead)
{
    __shared__ float red[256];
    __shared__ float ys[256];
    const int t = blockIdx.x, h = blockIdx.y, d = threadIdx.x;

    float x = src[(size_t)t * sStride + h * sHead + d];
    red[d] = x * x;
    __syncthreads();
    for (int s2 = 128; s2 > 0; s2 >>= 1) {
        if (d < s2) red[d] += red[d + s2];
        __syncthreads();
    }
    float rms = rsqrtf(red[0] * (1.0f / 256.0f) + 1e-6f);
    float y = x * rms * (1.0f + w[d]);
    ys[d] = y;
    __syncthreads();

    unsigned short* drow = dst + (size_t)t * dStride + h * dHead;
    if (d < 32) {
        float x1 = ys[d], x2 = ys[d + 32];
        // inv_freq = THETA^(-d/32) = exp(-d * ln(1e7)/32)
        float invf = expf(-(float)d * 0.50369048909583194f);
        float fr = (float)pos[t] * invf;
        float c = cosf(fr), s = sinf(fr);
        drow[d]      = f2bf(x1 * c - x2 * s);
        drow[d + 32] = f2bf(x2 * c + x1 * s);
    } else if (d >= 64) {
        drow[d] = f2bf(y);
    }
}

// ---------------------------------------------------------------------------
// Tiled transpose with f32->bf16: src[rows][cols] f32 -> dst[cols][rows] bf16.
// 32x32 tiles, 256 threads.
// ---------------------------------------------------------------------------
__global__ __launch_bounds__(256) void transpose_bf16(
    const float* __restrict__ src, unsigned short* __restrict__ dst,
    int rows, int cols)
{
    __shared__ unsigned short tile[32][33];
    const int t0 = blockIdx.x * 32, c0 = blockIdx.y * 32;
    const int tx = threadIdx.x & 31, ty = threadIdx.x >> 5;
    for (int j = 0; j < 4; ++j) {
        int r = ty + j * 8;
        tile[r][tx] = f2bf(src[(size_t)(t0 + r) * cols + c0 + tx]);
    }
    __syncthreads();
    for (int j = 0; j < 4; ++j) {
        int r = ty + j * 8;
        dst[(size_t)(c0 + r) * rows + t0 + tx] = tile[tx][r];
    }
}

// ---------------------------------------------------------------------------
// Flash attention, causal, GQA (H=16 over KV=2), D=256.
// Grid: (T/64, H). Block: 128 threads = 4 waves; each wave owns 16 q rows.
// Q fragments kept in registers for the whole key loop. K / V^T tiles of 32
// keys staged in LDS via CDNA5 async global->LDS b128 copies (ASYNCcnt).
// Online softmax with wave32 half-butterfly reductions; epilogue fuses
// 1/rowsum and sigmoid gating.
// ---------------------------------------------------------------------------
#define AQ 64
#define AK 32
#define HD 256
#define ATT_SCALE 0.0625f

__global__ __launch_bounds__(128) void attn(
    const unsigned short* __restrict__ qn,   // [T][H*D] bf16 (normed+roped)
    const unsigned short* __restrict__ kn,   // [T][KV*D] bf16 (normed+roped)
    const unsigned short* __restrict__ vtb,  // [KV*D][T] bf16 (V transposed)
    const float* __restrict__ qout,          // [T][H*2D] f32 (gate at h*512+256)
    float* __restrict__ obuf)                // [T][H*D] f32
{
    __shared__ alignas(16) unsigned short lK[AK * HD];    // [key][d]
    __shared__ alignas(16) unsigned short lVT[HD * AK];   // [d][key]
    __shared__ alignas(16) unsigned short lP[4][16 * AK]; // per-wave P [m][key]

    const int tid = threadIdx.x, lane = tid & 31, wave = tid >> 5;
    const int qt = blockIdx.x, h = blockIdx.y;
    const int kvh = h >> 3;                       // H / KV = 8
    const int g = lane >> 4, mr = lane & 15, nc = lane & 15;
    const int m0 = qt * AQ + wave * 16;

    // Load Q fragments (held in registers across the key loop)
    Frag qf[8];
    for (int f = 0; f < 8; ++f)
        for (int v = 0; v < 8; ++v) {
            int kb = f * 32 + (v & 3) * 2 + (v >> 2) * 16 + g * 8;
            qf[f].u[v] = *(const unsigned int*)&qn[(size_t)(m0 + mr) * 4096 + h * 256 + kb];
        }

    v8f accO[16] = {};
    float rm[8], rs[8];
    for (int r = 0; r < 8; ++r) { rm[r] = -3.0e38f; rs[r] = 0.0f; }

    const unsigned lK_base  = lds_off(&lK[0]);
    const unsigned lVT_base = lds_off(&lVT[0]);

    const int nkt = qt * 2 + 2;                   // causal key-tile bound (uniform per block)
    for (int kt = 0; kt < nkt; ++kt) {
        const int k0 = kt * AK;
        __syncthreads();
        // Async-stage K tile: 32 keys x 256 d bf16 = 1024 x 16B chunks, 8/thread
        for (int i = 0; i < 8; ++i) {
            int ch = tid + i * 128;
            int key = ch >> 5, c = ch & 31;       // 32 chunks per 512B key row
            async_copy16(lK_base + (unsigned)(key * HD * 2 + c * 16),
                         (const char*)kn + ((size_t)(k0 + key) * 512 + kvh * 256) * 2 + c * 16);
        }
        // Async-stage V^T tile: 256 d x 32 keys bf16 = 1024 x 16B chunks, 8/thread
        for (int i = 0; i < 8; ++i) {
            int ch = tid + i * 128;
            int d = ch >> 2, q = ch & 3;          // 4 chunks per 64B d row
            async_copy16(lVT_base + (unsigned)(d * AK * 2 + q * 16),
                         (const char*)vtb + ((size_t)(kvh * 256 + d) * 2048 + k0) * 2 + q * 16);
        }
        async_wait0();
        __syncthreads();

        // S = Q * K^T   (two 16-key N-subtiles)
        v8f s[2] = {};
        for (int nt = 0; nt < 2; ++nt)
            for (int f = 0; f < 8; ++f) {
                Frag b;
                for (int v = 0; v < 8; ++v) {
                    int kb = f * 32 + (v & 3) * 2 + (v >> 2) * 16 + g * 8;
                    b.u[v] = *(const unsigned int*)&lK[(nt * 16 + nc) * HD + kb];
                }
                s[nt] = __builtin_amdgcn_wmma_f32_16x16x32_bf16(
                    false, qf[f].v, false, b.v, (short)0, s[nt], false, false);
            }

        // Online softmax update (per accumulator row r; m = m0 + r + 8*g)
        for (int r = 0; r < 8; ++r) {
            int m = m0 + r + 8 * g;
            float s0 = s[0][r] * ATT_SCALE;
            float s1 = s[1][r] * ATT_SCALE;
            if (k0 + nc > m)      s0 = -3.0e38f;
            if (k0 + 16 + nc > m) s1 = -3.0e38f;
            float loc = fmaxf(s0, s1);
            for (int x = 1; x < 16; x <<= 1) loc = fmaxf(loc, __shfl_xor(loc, x, 32));
            float mn = fmaxf(rm[r], loc);
            float al = __expf(rm[r] - mn);
            rm[r] = mn;
            float p0 = __expf(s0 - mn);
            float p1 = __expf(s1 - mn);
            float sm = p0 + p1;
            for (int x = 1; x < 16; x <<= 1) sm += __shfl_xor(sm, x, 32);
            rs[r] = rs[r] * al + sm;
            for (int dt = 0; dt < 16; ++dt) accO[dt][r] *= al;
            lP[wave][(r + 8 * g) * AK + nc]      = f2bf(p0);
            lP[wave][(r + 8 * g) * AK + 16 + nc] = f2bf(p1);
        }
        __syncthreads();

        // O += P * V
        Frag pf;
        for (int v = 0; v < 8; ++v) {
            int kb = (v & 3) * 2 + (v >> 2) * 16 + g * 8;
            pf.u[v] = *(const unsigned int*)&lP[wave][mr * AK + kb];
        }
        for (int dt = 0; dt < 16; ++dt) {
            Frag b;
            for (int v = 0; v < 8; ++v) {
                int kb = (v & 3) * 2 + (v >> 2) * 16 + g * 8;
                b.u[v] = *(const unsigned int*)&lVT[(dt * 16 + nc) * AK + kb];
            }
            accO[dt] = __builtin_amdgcn_wmma_f32_16x16x32_bf16(
                false, pf.v, false, b.v, (short)0, accO[dt], false, false);
        }
    }

    // Epilogue: normalize + sigmoid gate
    for (int dt = 0; dt < 16; ++dt)
        for (int r = 0; r < 8; ++r) {
            int m = m0 + r + 8 * g;
            int d = dt * 16 + nc;
            float gate = qout[(size_t)m * 8192 + h * 512 + 256 + d];
            float o = (accO[dt][r] / rs[r]) * (1.0f / (1.0f + __expf(-gate)));
            obuf[(size_t)m * 4096 + h * 256 + d] = o;
        }
}

// ---------------------------------------------------------------------------
// Launch
// ---------------------------------------------------------------------------
extern "C" void kernel_launch(void* const* d_in, const int* in_sizes, int n_in,
                              void* d_out, int out_size, void* d_ws, size_t ws_size,
                              hipStream_t stream)
{
    const int*   positions = (const int*)d_in[0];
    const float* hidden    = (const float*)d_in[1];
    const float* wq        = (const float*)d_in[2];
    const float* wk        = (const float*)d_in[3];
    const float* wv        = (const float*)d_in[4];
    const float* wo        = (const float*)d_in[5];
    const float* qnw       = (const float*)d_in[6];
    const float* knw       = (const float*)d_in[7];
    float*       out       = (float*)d_out;

    char* ws = (char*)d_ws;
    float*          qoutb = (float*)ws;          ws += (size_t)2048 * 8192 * 4;  // 64 MB
    float*          kbuf  = (float*)ws;          ws += (size_t)2048 * 512 * 4;   // 4 MB
    float*          vbuf  = (float*)ws;          ws += (size_t)2048 * 512 * 4;   // 4 MB
    unsigned short* qn    = (unsigned short*)ws; ws += (size_t)2048 * 4096 * 2;  // 16 MB
    unsigned short* kn    = (unsigned short*)ws; ws += (size_t)2048 * 512 * 2;   // 2 MB
    unsigned short* vtb   = (unsigned short*)ws; ws += (size_t)512 * 2048 * 2;   // 2 MB
    float*          obuf  = (float*)ws;          ws += (size_t)2048 * 4096 * 4;  // 32 MB

    // QKV projections (bf16 WMMA, f32 accumulate)
    gemm_xwT<<<dim3(16, 128), 256, 0, stream>>>(hidden, wq, qoutb, 2048, 8192);
    gemm_xwT<<<dim3(16, 8),   256, 0, stream>>>(hidden, wk, kbuf,  2048, 512);
    gemm_xwT<<<dim3(16, 8),   256, 0, stream>>>(hidden, wv, vbuf,  2048, 512);

    // RMSNorm + RoPE, emit bf16 Q/K
    norm_rope<<<dim3(2048, 16), 256, 0, stream>>>(positions, qoutb, qnw, qn, 8192, 512, 4096, 256);
    norm_rope<<<dim3(2048, 2),  256, 0, stream>>>(positions, kbuf,  knw, kn, 512,  256, 512,  256);

    // V -> bf16, transposed to [KV*D][T] for contiguous async V^T-tile staging
    transpose_bf16<<<dim3(64, 16), 256, 0, stream>>>(vbuf, vtb, 2048, 512);

    // Causal flash attention + gating (async-LDS staged K/V^T)
    attn<<<dim3(32, 16), 128, 0, stream>>>(qn, kn, vtb, qoutb, obuf);

    // Output projection
    gemm_xwT<<<dim3(16, 32), 256, 0, stream>>>(obuf, wo, out, 4096, 2048);
}
